// RSSM_3178275799566
// MI455X (gfx1250) — matrix operand order repaired and checked by reading, not compiled, not run on hardware
//
#include <hip/hip_runtime.h>

// ---------------------------------------------------------------------------
// RSSM scan for MI455X (gfx1250): bf16 WMMA GEMMs with split-K (8 waves/tile,
// LDS reduction), uniform (scalar) K-loop so the compiler pipelines
// global_load_b128 clauses against back-to-back v_wmma. Weights pre-transposed
// to bf16 once per launch so the 64-step scan streams them from the 192MB L2.
// ---------------------------------------------------------------------------

#define BB   16
#define TT   64
#define AA   6
#define EE   12288
#define DET  4096
#define HID  1024
#define SD   1024   // S*D
#define K1P  1280   // img1 K (1030) padded to multiple of WPT*32 = 256
#define WPT  8      // waves per output tile (split-K factor)

typedef __bf16 bf16x8  __attribute__((ext_vector_type(8)));
typedef __bf16 v16bf   __attribute__((ext_vector_type(16)));
typedef float  v8f     __attribute__((ext_vector_type(8)));

static __device__ __forceinline__ __bf16 f2bf(float x) {
  unsigned u = __float_as_uint(x);
  unsigned r = (u + 0x7FFFu + ((u >> 16) & 1u)) >> 16;
  unsigned short h = (unsigned short)r;
  __bf16 o;
  __builtin_memcpy(&o, &h, 2);
  return o;
}

static __device__ __forceinline__ float sigm(float x) { return 1.0f / (1.0f + expf(-x)); }

// --------------------------- weight convert+transpose ----------------------
// W fp32 [K,N] row-major  ->  WT bf16 [N,Kpad] row-major (zero padded K tail)
__global__ void convT_kernel(const float* __restrict__ W, __bf16* __restrict__ WT,
                             int K, int N, int Kpad) {
  size_t idx = (size_t)blockIdx.x * blockDim.x + threadIdx.x;
  size_t total = (size_t)N * (size_t)Kpad;
  if (idx >= total) return;
  size_t n = idx / (size_t)Kpad;
  size_t k = idx % (size_t)Kpad;
  WT[idx] = (k < (size_t)K) ? f2bf(W[k * (size_t)N + n]) : f2bf(0.0f);
}

// --------------------------- init deter ------------------------------------
__global__ void tanh_init_kernel(const float* __restrict__ w_init,
                                 float* __restrict__ init_deter,
                                 __bf16* __restrict__ init_deter_bf) {
  int idx = blockIdx.x * blockDim.x + threadIdx.x;   // 16*4096
  int i = idx & (DET - 1);
  float v = tanhf(w_init[i]);
  init_deter[idx] = v;
  init_deter_bf[idx] = f2bf(v);
}

// --------------------------- WMMA GEMM (split-K) ---------------------------
// C[16,N] = A[16,K](bf16, lda) @ BT^T, BT bf16 [N, ldb] pre-transposed.
// One block (8 waves) per 16x16 output tile; wave w accumulates the w-th
// contiguous K/8 chunk with v_wmma_f32_16x16x32_bf16; partials reduced in LDS.
// K must be a multiple of 256 so the loop trip count (K>>8) is wave-uniform:
// scalar loop control -> compiler unrolls & pipelines loads vs WMMA.
__global__ __launch_bounds__(256) void gemm_wmma_bf16(
    const __bf16* __restrict__ A, int lda,
    const __bf16* __restrict__ BT, int ldb,
    const float* __restrict__ bias,
    float* __restrict__ C, int ldc, int K) {
  __shared__ float red[WPT * 256];
  const int tid  = threadIdx.x;
  const int lane = tid & 31;
  const int wave = tid >> 5;          // 0..7: which K-chunk
  const int ncol = blockIdx.x * 16;
  const int half = lane >> 4;         // 0: lanes 0-15, 1: lanes 16-31
  const int r    = lane & 15;

  const int steps = K >> 8;           // uniform 32-wide steps per wave

  // thread-varying base pointers; wave K-offset folded in here
  const __bf16* ap = A  + (size_t)r * lda + half * 8  + (size_t)wave * steps * 32;
  const __bf16* bp = BT + (size_t)(ncol + r) * ldb + half * 8  + (size_t)wave * steps * 32;

  v8f acc = {0.f, 0.f, 0.f, 0.f, 0.f, 0.f, 0.f, 0.f};
#pragma unroll 4
  for (int s = 0; s < steps; ++s) {
    int k0 = s * 32;
    // A fragment 16x32: lane half selects K {0..7,16..23} or {8..15,24..31}
    bf16x8 a0 = *(const bf16x8*)(ap + k0);
    bf16x8 a1 = *(const bf16x8*)(ap + k0 + 16);
    // B fragment 32x16: lane half selects K {0..7,16..23} or {8..15,24..31}
    bf16x8 b0 = *(const bf16x8*)(bp + k0);
    bf16x8 b1 = *(const bf16x8*)(bp + k0 + 16);
    v16bf av = __builtin_shufflevector(a0, a1, 0,1,2,3,4,5,6,7,8,9,10,11,12,13,14,15);
    v16bf bv = __builtin_shufflevector(b0, b1, 0,1,2,3,4,5,6,7,8,9,10,11,12,13,14,15);
    acc = __builtin_amdgcn_wmma_f32_16x16x32_bf16(false, av, false, bv,
                                                  (short)0, acc, false, false);
  }
  // stash partials: slot (wave, r2, lane)
#pragma unroll
  for (int r2 = 0; r2 < 8; ++r2)
    red[wave * 256 + r2 * 32 + lane] = acc[r2];
  __syncthreads();
  // each of the 256 threads owns one (r2, lane) slot; sum the 8 partials
  int rr = tid >> 5;
  int ll = tid & 31;
  float s = 0.f;
#pragma unroll
  for (int w = 0; w < WPT; ++w) s += red[w * 256 + rr * 32 + ll];
  int m = rr + (ll >> 4) * 8;          // D layout: VGPR rr -> row rr (+8)
  int c = ll & 15;
  float bb = bias ? bias[ncol + c] : 0.0f;
  C[(size_t)m * ldc + ncol + c] = s + bb;
}

// --------------------------- LayerNorm (+ optional SiLU) -------------------
// One block per row. outF may alias X (in-place). outB optional bf16 sink.
__global__ __launch_bounds__(256) void ln_act_kernel(
    const float* __restrict__ X, int n,
    const float* __restrict__ g, const float* __restrict__ b,
    float* outF, __bf16* outB, int ldB, int offB, int act) {
  __shared__ float s1s[256], s2s[256];
  int row = blockIdx.x;
  const float* x = X + (size_t)row * n;
  float s1 = 0.f, s2 = 0.f;
  for (int i = threadIdx.x; i < n; i += 256) { float v = x[i]; s1 += v; s2 += v * v; }
  s1s[threadIdx.x] = s1; s2s[threadIdx.x] = s2;
  __syncthreads();
  for (int off = 128; off > 0; off >>= 1) {
    if (threadIdx.x < off) { s1s[threadIdx.x] += s1s[threadIdx.x + off];
                             s2s[threadIdx.x] += s2s[threadIdx.x + off]; }
    __syncthreads();
  }
  float mean = s1s[0] / n;
  float var  = s2s[0] / n - mean * mean;
  float rs   = rsqrtf(var + 1e-3f);
  for (int i = threadIdx.x; i < n; i += 256) {
    float y = (x[i] - mean) * rs * g[i] + b[i];
    if (act) y = y * sigm(y);
    if (outF) outF[(size_t)row * n + i] = y;
    if (outB) outB[(size_t)row * ldB + offB + i] = f2bf(y);
  }
}

// --------------------------- per-step prologue -----------------------------
// Masks carry with is_first, clips action, builds bf16 GEMM inputs.
__global__ __launch_bounds__(256) void prep_kernel(
    const float* __restrict__ stoch, float* __restrict__ deter,
    const float* __restrict__ init_stoch, const float* __restrict__ init_deter,
    const float* __restrict__ act_base,   // prev_actions + t*A   (stride T*A per b)
    const float* __restrict__ firsts,     // is_firsts + t        (stride T per b)
    __bf16* __restrict__ xin1,            // [16,1280]  stoch ++ clipped a ++ pad
    __bf16* __restrict__ xin2) {          // [16,5120]  (x filled later) ++ deter
  int idx = blockIdx.x * blockDim.x + threadIdx.x;  // 16*4096
  int b = idx >> 12, i = idx & (DET - 1);
  float f = firsts[(size_t)b * TT];
  float nf = 1.0f - f;
  float dm = deter[(size_t)b * DET + i] * nf + init_deter[(size_t)b * DET + i] * f;
  deter[(size_t)b * DET + i] = dm;
  xin2[(size_t)b * (HID + DET) + HID + i] = f2bf(dm);
  if (i < SD) {
    float sm = stoch[(size_t)b * SD + i] * nf + init_stoch[(size_t)b * SD + i] * f;
    xin1[(size_t)b * K1P + i] = f2bf(sm);
  }
  if (i >= SD && i < K1P) {
    int j = i - SD;
    float v = 0.0f;
    if (j < AA) {
      float a = act_base[(size_t)b * TT * AA + j];
      a = a / fmaxf(fabsf(a), 1.0f);   // clip_act (idempotent)
      v = a * nf;
    }
    xin1[(size_t)b * K1P + i] = f2bf(v);
  }
}

// --------------------------- GRU gate math ---------------------------------
__global__ __launch_bounds__(256) void gru_kernel(
    const float* __restrict__ parts,   // LN'd [16,12288]
    float* __restrict__ deter,         // in: masked deter, out: deter_new
    __bf16* __restrict__ deter_bf,     // [16,4096] GEMM3 input
    __bf16* __restrict__ xin5,         // [16,16384], deter goes at offset 12288
    float* __restrict__ out_deter) {   // d_out + t*8192 + 1024
  int idx = blockIdx.x * blockDim.x + threadIdx.x;  // 16*4096
  int b = idx >> 12, i = idx & (DET - 1);
  const float* p = parts + (size_t)b * (3 * DET);
  float r = sigm(p[i]);
  float c = tanhf(r * p[DET + i]);
  float u = sigm(p[2 * DET + i] - 1.0f);
  float d = deter[(size_t)b * DET + i];
  float dn = u * c + (1.0f - u) * d;
  deter[(size_t)b * DET + i] = dn;
  deter_bf[(size_t)b * DET + i] = f2bf(dn);
  xin5[(size_t)b * (EE + DET) + EE + i] = f2bf(dn);
  out_deter[(size_t)b * TT * 8192 + i] = dn;
}

// --------------------------- embed -> bf16 ---------------------------------
__global__ __launch_bounds__(256) void embed_kernel(const float* __restrict__ emb, // embed + t*E
                                                    __bf16* __restrict__ xin5) {
  int idx = blockIdx.x * blockDim.x + threadIdx.x;  // 16*12288
  int b = idx / EE, j = idx - b * EE;
  xin5[(size_t)b * (EE + DET) + j] = f2bf(emb[(size_t)b * TT * EE + j]);
}

// --------------------------- ST gumbel / mode sample -----------------------
// Forward value of oh + p - sg(p) is exactly the one-hot. One warp per (b,s).
__global__ __launch_bounds__(256) void sample_kernel(
    const float* __restrict__ logits,  // [16,1024]
    const float* __restrict__ gum,     // nullable, pre-offset by t (stride T*1024 per b)
    float* stoch_carry,                // nullable [16,1024]
    float* out_oh,                     // nullable, d_out + t*8192 + col
    float* out_lg,                     // nullable, d_out + t*8192 + col
    int useGum) {
  int lane = threadIdx.x & 31;
  int row  = blockIdx.x * 8 + (threadIdx.x >> 5);   // 512 rows = (b,s)
  int b = row >> 5, s = row & 31;
  int li = b * SD + s * 32 + lane;
  float l = logits[li];
  float m = l;
  for (int o = 16; o; o >>= 1) m = fmaxf(m, __shfl_xor(m, o, 32));
  float e = expf(l - m);
  float sum = e;
  for (int o = 16; o; o >>= 1) sum += __shfl_xor(sum, o, 32);
  float pm = 0.99f * (e / sum) + 0.01f / 32.0f;     // (1-UM)*p + UM/D
  float z = useGum ? (logf(pm) + gum[(size_t)b * TT * SD + s * 32 + lane]) : pm;
  float bv = z; int bi = lane;                      // argmax, first-index ties
  for (int o = 16; o; o >>= 1) {
    float ov = __shfl_xor(bv, o, 32);
    int   oi = __shfl_xor(bi, o, 32);
    if (ov > bv || (ov == bv && oi < bi)) { bv = ov; bi = oi; }
  }
  float oh = (lane == bi) ? 1.0f : 0.0f;
  if (stoch_carry) stoch_carry[li] = oh;
  size_t ob = (size_t)b * TT * 8192 + s * 32 + lane;
  if (out_oh) out_oh[ob] = oh;
  if (out_lg) out_lg[ob] = l;
}

// ===========================================================================
extern "C" void kernel_launch(void* const* d_in, const int* in_sizes, int n_in,
                              void* d_out, int out_size, void* d_ws, size_t ws_size,
                              hipStream_t stream) {
  const float* embed     = (const float*)d_in[0];
  const float* prev_act  = (const float*)d_in[1];
  const float* is_firsts = (const float*)d_in[2];
  const float* gum_prior = (const float*)d_in[3];
  const float* gum_post  = (const float*)d_in[4];
  const float* w_init    = (const float*)d_in[5];
  const float* W_img1    = (const float*)d_in[6];
  const float* g_img1    = (const float*)d_in[7];
  const float* b_img1    = (const float*)d_in[8];
  const float* W_gru     = (const float*)d_in[9];
  const float* g_gru     = (const float*)d_in[10];
  const float* b_gru     = (const float*)d_in[11];
  const float* W_img2a   = (const float*)d_in[12];
  const float* g_img2    = (const float*)d_in[13];
  const float* b_img2    = (const float*)d_in[14];
  const float* W_img2b   = (const float*)d_in[15];
  const float* bb_img2   = (const float*)d_in[16];
  const float* W_obs     = (const float*)d_in[17];
  const float* g_obs     = (const float*)d_in[18];
  const float* b_obs     = (const float*)d_in[19];
  const float* W_obsb    = (const float*)d_in[20];
  const float* bb_obs    = (const float*)d_in[21];
  float* out = (float*)d_out;

  // ---- workspace carve (256B aligned) ----
  char* p = (char*)d_ws;
  auto carve = [&](size_t bytes) { char* q = p; p += (bytes + 255) & ~(size_t)255; return q; };
  __bf16* wt_gru   = (__bf16*)carve((size_t)12288 * 5120 * 2);
  __bf16* wt_obs   = (__bf16*)carve((size_t)1024 * 16384 * 2);
  __bf16* wt_img2a = (__bf16*)carve((size_t)1024 * 4096 * 2);
  __bf16* wt_img1  = (__bf16*)carve((size_t)1024 * K1P * 2);
  __bf16* wt_img2b = (__bf16*)carve((size_t)1024 * 1024 * 2);
  __bf16* wt_obsb  = (__bf16*)carve((size_t)1024 * 1024 * 2);
  float*  init_deter    = (float*)carve((size_t)BB * DET * 4);
  __bf16* init_deter_bf = (__bf16*)carve((size_t)BB * DET * 2);
  float*  init_stoch    = (float*)carve((size_t)BB * SD * 4);
  float*  stoch  = (float*)carve((size_t)BB * SD * 4);
  float*  deter  = (float*)carve((size_t)BB * DET * 4);
  __bf16* deter_bf = (__bf16*)carve((size_t)BB * DET * 2);
  __bf16* xin1   = (__bf16*)carve((size_t)BB * K1P * 2);
  __bf16* xin2   = (__bf16*)carve((size_t)BB * (HID + DET) * 2);
  __bf16* xin5   = (__bf16*)carve((size_t)BB * (EE + DET) * 2);
  float*  y1     = (float*)carve((size_t)BB * HID * 4);
  float*  parts  = (float*)carve((size_t)BB * 3 * DET * 4);
  float*  hraw   = (float*)carve((size_t)BB * HID * 4);
  __bf16* h_bf   = (__bf16*)carve((size_t)BB * HID * 2);
  float*  lp     = (float*)carve((size_t)BB * SD * 4);
  float*  h2raw  = (float*)carve((size_t)BB * HID * 4);
  __bf16* h2_bf  = (__bf16*)carve((size_t)BB * HID * 2);
  float*  lq     = (float*)carve((size_t)BB * SD * 4);
  (void)ws_size; (void)in_sizes; (void)n_in; (void)out_size;

  auto conv = [&](const float* W, __bf16* WT, int K, int N, int Kpad) {
    size_t total = (size_t)N * Kpad;
    convT_kernel<<<(unsigned)((total + 255) / 256), 256, 0, stream>>>(W, WT, K, N, Kpad);
  };
  // Once per launch: fp32 [K,N] -> bf16 [N,Kpad]; all six then live in L2 (174MB < 192MB).
  conv(W_gru,   wt_gru,   5120, 12288, 5120);
  conv(W_obs,   wt_obs,   16384, 1024, 16384);
  conv(W_img2a, wt_img2a, 4096,  1024, 4096);
  conv(W_img1,  wt_img1,  1030,  1024, K1P);
  conv(W_img2b, wt_img2b, 1024,  1024, 1024);
  conv(W_obsb,  wt_obsb,  1024,  1024, 1024);

  const dim3 gN1(64);    // N=1024  -> 64 tiles, one block (8 waves) per tile
  const dim3 gN12(768);  // N=12288 -> 768 tiles
  // ---- init state: init_deter = tanh(w_init); init_stoch = st_mode(prior(init_deter))
  tanh_init_kernel<<<(BB * DET) / 256, 256, 0, stream>>>(w_init, init_deter, init_deter_bf);
  gemm_wmma_bf16<<<gN1, 256, 0, stream>>>(init_deter_bf, DET, wt_img2a, 4096, nullptr, hraw, HID, 4096);
  ln_act_kernel<<<BB, 256, 0, stream>>>(hraw, HID, g_img2, b_img2, nullptr, h_bf, HID, 0, 1);
  gemm_wmma_bf16<<<gN1, 256, 0, stream>>>(h_bf, HID, wt_img2b, 1024, bb_img2, lp, SD, 1024);
  sample_kernel<<<64, 256, 0, stream>>>(lp, nullptr, init_stoch, nullptr, nullptr, 0);

  for (int t = 0; t < TT; ++t) {
    float* outp = out + (size_t)t * 8192;
    // prologue: mask carry with is_first, clip action, build bf16 inputs
    prep_kernel<<<(BB * DET) / 256, 256, 0, stream>>>(
        stoch, deter, init_stoch, init_deter,
        prev_act + (size_t)t * AA, is_firsts + t, xin1, xin2);
    // x = silu(LN([stoch,a] @ W_img1))
    gemm_wmma_bf16<<<gN1, 256, 0, stream>>>(xin1, K1P, wt_img1, K1P, nullptr, y1, HID, K1P);
    ln_act_kernel<<<BB, 256, 0, stream>>>(y1, HID, g_img1, b_img1, nullptr, xin2, HID + DET, 0, 1);
    // parts = LN([x,deter] @ W_gru); GRU update
    gemm_wmma_bf16<<<gN12, 256, 0, stream>>>(xin2, HID + DET, wt_gru, 5120, nullptr, parts, 3 * DET, 5120);
    ln_act_kernel<<<BB, 256, 0, stream>>>(parts, 3 * DET, g_gru, b_gru, parts, nullptr, 0, 0, 0);
    gru_kernel<<<(BB * DET) / 256, 256, 0, stream>>>(parts, deter, deter_bf, xin5, outp + 1024);
    // prior logits + sample
    gemm_wmma_bf16<<<gN1, 256, 0, stream>>>(deter_bf, DET, wt_img2a, 4096, nullptr, hraw, HID, 4096);
    ln_act_kernel<<<BB, 256, 0, stream>>>(hraw, HID, g_img2, b_img2, nullptr, h_bf, HID, 0, 1);
    gemm_wmma_bf16<<<gN1, 256, 0, stream>>>(h_bf, HID, wt_img2b, 1024, bb_img2, lp, SD, 1024);
    sample_kernel<<<64, 256, 0, stream>>>(lp, gum_prior + (size_t)t * SD, nullptr,
                                          outp + 6144, outp + 7168, 1);
    // posterior: [embed, deter_new] @ W_obs -> logits -> sample
    embed_kernel<<<(BB * EE) / 256, 256, 0, stream>>>(embed + (size_t)t * EE, xin5);
    gemm_wmma_bf16<<<gN1, 256, 0, stream>>>(xin5, EE + DET, wt_obs, 16384, nullptr, h2raw, HID, 16384);
    ln_act_kernel<<<BB, 256, 0, stream>>>(h2raw, HID, g_obs, b_obs, nullptr, h2_bf, HID, 0, 1);
    gemm_wmma_bf16<<<gN1, 256, 0, stream>>>(h2_bf, HID, wt_obsb, 1024, bb_obs, lq, SD, 1024);
    sample_kernel<<<64, 256, 0, stream>>>(lq, gum_post + (size_t)t * SD, stoch,
                                          outp + 0, outp + 5120, 1);
  }
}
